// Encoder_8993661518585
// MI455X (gfx1250) — matrix operand order repaired and checked by reading, not compiled, not run on hardware
//
#include <hip/hip_runtime.h>

typedef __attribute__((ext_vector_type(16))) _Float16 v16h;
typedef __attribute__((ext_vector_type(8)))  _Float16 v8h;
typedef __attribute__((ext_vector_type(8)))  float    v8f;

#define T_STEPS 64
#define D_SZ    32
#define HID     128
#define WID     128
#define NTHREADS 256

// Native CDNA5 tanh (V_TANH_F32, single TRANS32 op) when available
#if __has_builtin(__builtin_amdgcn_tanhf)
#define FAST_TANH(x) __builtin_amdgcn_tanhf(x)
#else
#define FAST_TANH(x) tanhf(x)
#endif

// Tsit5 tableau
#define C_A21 0.161f
#define C_A31 -0.008480655492356989f
#define C_A32 0.335480655492357f
#define C_A41 2.8971530571054935f
#define C_A42 -6.359448489975075f
#define C_A43 4.3622954328695815f
#define C_A51 5.325864828439257f
#define C_A52 -11.748883564062828f
#define C_A53 7.4955393428898365f
#define C_A54 -0.09249506636175525f
#define C_A61 5.86145544294642f
#define C_A62 -12.92096931784711f
#define C_A63 8.159367898576159f
#define C_A64 -0.071584973281401f
#define C_A65 -0.028269050394068383f
#define C_B1 0.09646076681806523f
#define C_B2 0.01f
#define C_B3 0.4798896504144996f
#define C_B4 1.379008574103742f
#define C_B5 -3.290069515436081f
#define C_B6 2.324710524099774f

// ---- dynamic LDS layout (bytes) ----
#define OFF_Y      0         //  8 KB  float  y[16][128]
#define OFF_K      8192      // 48 KB  float  k[6][16][128]
#define OFF_G      57344     //  2 KB  float  g[16][32]
#define OFF_S0     59392     //  4 KB  f16 staging
#define OFF_S1     63488     //  4 KB
#define OFF_S2     67584     //  4 KB
#define OFF_V0L    71680     // 32 KB  f16 V0 (LDS-resident)
#define OFF_V1L    104448    // 32 KB  f16 V1 (LDS-resident)
#define OFF_VST    137216    // 64 KB  per-wave V2 double-buffers (8 x 2 x 4KB)
#define SMEM_BYTES 202752

// ---------------------------------------------------------------------------
// WMMA fragment loaders (CDNA5 16x16x32 f16, wave32)
// A (16x32, 16-bit): lane L holds row m=L%16; 16 contiguous K values
// starting at kb+16*hf (same K order as B)
// B (32x16, 16-bit): lane L holds column n=L%16; 16 contiguous K values
// starting at kb+16*hf. Weight stored row-major [N][K] == B column-major.
// ---------------------------------------------------------------------------
__device__ __forceinline__ v16h load_a_frag(const _Float16* __restrict__ s,
                                            int lane, int kb) {
  const int m  = lane & 15;
  const int hf = lane >> 4;
  const v8h lo = *(const v8h*)(s + m * WID + kb + 16 * hf);
  const v8h hi = *(const v8h*)(s + m * WID + kb + 16 * hf + 8);
  v16h r;
#pragma unroll
  for (int i = 0; i < 8; ++i) { r[i] = lo[i]; r[i + 8] = hi[i]; }
  return r;
}

__device__ __forceinline__ v16h load_b_frag(const _Float16* __restrict__ w,
                                            int kstride, int n, int lane, int kb) {
  const int hf = lane >> 4;
  return *(const v16h*)(w + (size_t)n * kstride + kb + 16 * hf);
}

// 16xN(=128) x K matmul: one 16-col tile per wave (8 waves cover N=128).
// MODE: 0 = identity, 1 = relu, 2 = tanh.  Writes f16 staging, optional f32.
template <int K, int MODE>
__device__ __forceinline__ void mm_layer(const _Float16* __restrict__ inS,
                                         const _Float16* __restrict__ wg,
                                         const float* __restrict__ bias,
                                         _Float16* __restrict__ outS,
                                         float* __restrict__ outF,
                                         int lane, int wave) {
  const int l15 = lane & 15;
  const int hf  = lane >> 4;
  const int n   = wave * 16 + l15;
  v8f c = {0.f, 0.f, 0.f, 0.f, 0.f, 0.f, 0.f, 0.f};
#pragma unroll
  for (int kt = 0; kt < K / 32; ++kt) {
    v16h a = load_a_frag(inS, lane, kt * 32);
    v16h b = load_b_frag(wg, K, n, lane, kt * 32);
    c = __builtin_amdgcn_wmma_f32_16x16x32_f16(false, a, false, b,
                                               (short)0, c, false, false);
  }
  const float bv = bias[n];
#pragma unroll
  for (int r = 0; r < 8; ++r) {
    float v = c[r] + bv;
    if (MODE == 1) v = v > 0.f ? v : 0.f;
    if (MODE == 2) v = FAST_TANH(v);
    const int m = r + 8 * hf;
    outS[m * WID + n] = (_Float16)v;
    if (outF) outF[m * HID + n] = v;
  }
}

// ---------------------------------------------------------------------------
// Async staging: copy one 16-row (4 KB, contiguous) V2 tile into a per-wave
// LDS buffer with CDNA5 async-to-LDS (tracked by ASYNCcnt).  Inline asm is
// used for toolchain portability; the instruction's IOFFSET applies to both
// the global and LDS addresses (ISA 10.4.4), so one base pair suffices.
// 8 x b128 per wave (32 lanes x 16B each) = 4 KB.
// ---------------------------------------------------------------------------
__device__ __forceinline__ void stage_tile_async(const _Float16* __restrict__ V2h,
                                                 const _Float16* __restrict__ vst,
                                                 int j, int buf, int lane) {
  const unsigned long long ga =
      (unsigned long long)(size_t)((const char*)(V2h + (size_t)j * 16 * WID) + lane * 16);
  const unsigned lo =
      (unsigned)(size_t)((const char*)(vst + buf * 2048) + lane * 16);
#pragma unroll
  for (int i = 0; i < 8; ++i) {
    asm volatile("global_load_async_to_lds_b128 %0, %1, off offset:%c2"
                 :: "v"(lo), "v"(ga), "i"(i * 512)
                 : "memory");
  }
}

// Final vector-field layer fused with the D-contraction:
// k[b,h] = sum_d tanh( (z @ V2^T + vb2)[b, h*32+d] ) * g[b,d]
// Wave w owns 32 consecutive 16-col tiles jj=0..31 (two per output h),
// double-buffered through LDS with a 2-deep async pipeline.
__device__ __forceinline__ void vf_contract(const _Float16* __restrict__ inS,
                                            const _Float16* __restrict__ V2h,
                                            const float* __restrict__ vb2,
                                            const float gv[2][8],
                                            float* __restrict__ kdst,
                                            const _Float16* __restrict__ vst,
                                            int lane, int wave) {
  const int l15 = lane & 15;
  const int hf  = lane >> 4;
  v16h afr[4];
#pragma unroll
  for (int kt = 0; kt < 4; ++kt) afr[kt] = load_a_frag(inS, lane, kt * 32);

  stage_tile_async(V2h, vst, 32 * wave + 0, 0, lane);
  stage_tile_async(V2h, vst, 32 * wave + 1, 1, lane);

  float acc[8] = {0.f, 0.f, 0.f, 0.f, 0.f, 0.f, 0.f, 0.f};
  for (int jj = 0; jj < 32; ++jj) {
    const int p = jj & 1;
    const int n = (32 * wave + jj) * 16 + l15;   // output column hd == V2 row

    // tile jj complete when at most the next tile (8 loads) is outstanding
    if (jj == 31) asm volatile("s_wait_asynccnt 0x0" ::: "memory");
    else          asm volatile("s_wait_asynccnt 0x8" ::: "memory");

    const _Float16* bsrc = vst + (jj & 1) * 2048 + l15 * 128 + 16 * hf;
    v8f c = {0.f, 0.f, 0.f, 0.f, 0.f, 0.f, 0.f, 0.f};
#pragma unroll
    for (int kt = 0; kt < 4; ++kt) {
      v16h b = *(const v16h*)(bsrc + kt * 32);
      c = __builtin_amdgcn_wmma_f32_16x16x32_f16(false, afr[kt], false, b,
                                                 (short)0, c, false, false);
    }
    if (jj + 2 < 32) {
      // make sure this buffer's DS reads landed before async writes reuse it
      asm volatile("s_wait_dscnt 0x0" ::: "memory");
      stage_tile_async(V2h, vst, 32 * wave + jj + 2, jj & 1, lane);
    }

    const float bv = vb2[n];
#pragma unroll
    for (int r = 0; r < 8; ++r) {
      float w = FAST_TANH(c[r] + bv) * gv[p][r];
      // reduce over the 16 columns (16-lane groups) of this tile
      w += __shfl_xor(w, 1, 16);
      w += __shfl_xor(w, 2, 16);
      w += __shfl_xor(w, 4, 16);
      w += __shfl_xor(w, 8, 16);
      acc[r] += w;
    }
    if (p == 1) {
      const int h = wave * 16 + (jj >> 1);
      if (l15 == 0) {
#pragma unroll
        for (int r = 0; r < 8; ++r) kdst[(r + 8 * hf) * HID + h] = acc[r];
      }
#pragma unroll
      for (int r = 0; r < 8; ++r) acc[r] = 0.f;
    }
  }
  asm volatile("s_wait_asynccnt 0x0" ::: "memory");  // drain before reuse
}

__device__ __forceinline__ void f_eval(const _Float16* S0, _Float16* S1, _Float16* S2,
                                       const _Float16* V0L, const float* vb0,
                                       const _Float16* V1L, const float* vb1,
                                       const _Float16* V2h, const float* vb2,
                                       const float gv[2][8], float* kdst,
                                       const _Float16* vst, int lane, int wave) {
  mm_layer<128, 2>(S0, V0L, vb0, S1, nullptr, lane, wave);
  __syncthreads();
  mm_layer<128, 2>(S1, V1L, vb1, S2, nullptr, lane, wave);
  __syncthreads();
  vf_contract(S2, V2h, vb2, gv, kdst, vst, lane, wave);
  __syncthreads();
}

// ytmp = y + hh * sum_j coef[j]*k_j  -> f16 staging (and optionally back to y)
template <int NK>
__device__ __forceinline__ void combine(const float* __restrict__ y_s,
                                        const float* __restrict__ ks,
                                        const float* coef, float hh,
                                        _Float16* __restrict__ S0,
                                        float* __restrict__ yout, int tid) {
  for (int i = tid; i < 16 * HID; i += NTHREADS) {
    float acc = 0.f;
#pragma unroll
    for (int j = 0; j < NK; ++j) acc += coef[j] * ks[j * 16 * HID + i];
    const float v = y_s[i] + hh * acc;
    S0[i] = (_Float16)v;
    if (yout) yout[i] = v;
  }
}

// ---------------------------------------------------------------------------
// f32 -> f16 weight conversion into workspace (one-time per call)
// Layout: W0[128*32] W1[128*128] W2[128*128] V0[128*128] V1[128*128] V2[4096*128]
// ---------------------------------------------------------------------------
#define N_W0 (128 * 32)
#define N_SQ (128 * 128)
#define N_V2 (4096 * 128)
#define N_TOTAL (N_W0 + 4 * N_SQ + N_V2)

__global__ void convert_weights(const float* __restrict__ W0, const float* __restrict__ W1,
                                const float* __restrict__ W2, const float* __restrict__ V0,
                                const float* __restrict__ V1, const float* __restrict__ V2,
                                _Float16* __restrict__ ws) {
  int i = blockIdx.x * blockDim.x + threadIdx.x;
  if (i >= N_TOTAL) return;
  int idx = i;
  if (idx < N_W0) { ws[i] = (_Float16)W0[idx]; return; } idx -= N_W0;
  if (idx < N_SQ) { ws[i] = (_Float16)W1[idx]; return; } idx -= N_SQ;
  if (idx < N_SQ) { ws[i] = (_Float16)W2[idx]; return; } idx -= N_SQ;
  if (idx < N_SQ) { ws[i] = (_Float16)V0[idx]; return; } idx -= N_SQ;
  if (idx < N_SQ) { ws[i] = (_Float16)V1[idx]; return; } idx -= N_SQ;
  ws[i] = (_Float16)V2[idx];
}

// ---------------------------------------------------------------------------
// Main persistent kernel: each workgroup owns 16 batch rows and runs the
// whole Tsit5 scan. 8 blocks x 256 threads (8 wave32s).
// ---------------------------------------------------------------------------
__global__ __launch_bounds__(NTHREADS) void cde_kernel(
    const float* __restrict__ ts, const float* __restrict__ coeffs,
    const float* __restrict__ b0, const float* __restrict__ b1,
    const float* __restrict__ b2, const float* __restrict__ vb0,
    const float* __restrict__ vb1, const float* __restrict__ vb2,
    const _Float16* __restrict__ wsh, const int* __restrict__ gcp,
    float* __restrict__ out) {
  extern __shared__ char smem[];
  float*    y_s = (float*)(smem + OFF_Y);
  float*    k_s = (float*)(smem + OFF_K);
  float*    g_s = (float*)(smem + OFF_G);
  _Float16* S0  = (_Float16*)(smem + OFF_S0);
  _Float16* S1  = (_Float16*)(smem + OFF_S1);
  _Float16* S2  = (_Float16*)(smem + OFF_S2);
  _Float16* V0L = (_Float16*)(smem + OFF_V0L);
  _Float16* V1L = (_Float16*)(smem + OFF_V1L);

  const int tid  = threadIdx.x;
  const int lane = tid & 31;
  const int wave = tid >> 5;
  const int bb   = blockIdx.x * 16;
  const int gc   = *gcp;

  const _Float16* W0h = wsh;
  const _Float16* W1h = W0h + N_W0;
  const _Float16* W2h = W1h + N_SQ;
  const _Float16* V0h = W2h + N_SQ;
  const _Float16* V1h = V0h + N_SQ;
  const _Float16* V2h = V1h + N_SQ;
  const _Float16* vst = (const _Float16*)(smem + OFF_VST) + wave * 4096;

  // ---- park V0/V1 (f16, 32 KB each) in LDS for the whole scan ----
  {
    const uint4* s0 = (const uint4*)V0h;  uint4* d0 = (uint4*)V0L;
    const uint4* s1 = (const uint4*)V1h;  uint4* d1 = (uint4*)V1L;
    for (int i = tid; i < 2048; i += NTHREADS) { d0[i] = s0[i]; d1[i] = s1[i]; }
  }

  // ---- initial MLP: y0 = W2(relu(W1(relu(W0 x + b0)) + b1)) + b2 ----
  for (int i = tid; i < 16 * D_SZ; i += NTHREADS) {
    const int m = i >> 5, d = i & 31;
    S0[m * WID + d] = (_Float16)coeffs[((size_t)(bb + m) * T_STEPS + 0) * D_SZ + d];
  }
  __syncthreads();
  mm_layer<32, 1>(S0, W0h, b0, S1, nullptr, lane, wave);
  __syncthreads();
  mm_layer<128, 1>(S1, W1h, b1, S2, nullptr, lane, wave);
  __syncthreads();
  mm_layer<128, 0>(S2, W2h, b2, S0, y_s, lane, wave);   // S0 <- f16(y), y_s <- y
  __syncthreads();

  if (gc) {
    for (int i = tid; i < 16 * HID; i += NTHREADS) {
      const int m = i >> 7, h = i & 127;
      out[((size_t)(bb + m) * T_STEPS + 0) * HID + h] = y_s[i];
    }
  }

  float gv[2][8];

  for (int t = 0; t < T_STEPS - 1; ++t) {
    const float dt = ts[t + 1] - ts[t];
    const float hh = dt * 0.5f;

    // piecewise-constant dX/dt for this interval
    for (int i = tid; i < 16 * D_SZ; i += NTHREADS) {
      const int m = i >> 5, d = i & 31;
      const float c1 = coeffs[((size_t)(bb + m) * T_STEPS + (t + 1)) * D_SZ + d];
      const float c0 = coeffs[((size_t)(bb + m) * T_STEPS + t) * D_SZ + d];
      g_s[i] = (c1 - c0) / dt;
    }
    __syncthreads();
    {
      const int l15 = lane & 15, hf = lane >> 4;
#pragma unroll
      for (int p = 0; p < 2; ++p)
#pragma unroll
        for (int r = 0; r < 8; ++r)
          gv[p][r] = g_s[(r + 8 * hf) * D_SZ + 16 * p + l15];
    }

    for (int sub = 0; sub < 2; ++sub) {
      // invariant: S0 holds f16(y) on entry
      f_eval(S0, S1, S2, V0L, vb0, V1L, vb1, V2h, vb2, gv, k_s + 0 * 2048, vst, lane, wave);
      { const float c[1] = {C_A21};
        combine<1>(y_s, k_s, c, hh, S0, nullptr, tid); }
      __syncthreads();
      f_eval(S0, S1, S2, V0L, vb0, V1L, vb1, V2h, vb2, gv, k_s + 1 * 2048, vst, lane, wave);
      { const float c[2] = {C_A31, C_A32};
        combine<2>(y_s, k_s, c, hh, S0, nullptr, tid); }
      __syncthreads();
      f_eval(S0, S1, S2, V0L, vb0, V1L, vb1, V2h, vb2, gv, k_s + 2 * 2048, vst, lane, wave);
      { const float c[3] = {C_A41, C_A42, C_A43};
        combine<3>(y_s, k_s, c, hh, S0, nullptr, tid); }
      __syncthreads();
      f_eval(S0, S1, S2, V0L, vb0, V1L, vb1, V2h, vb2, gv, k_s + 3 * 2048, vst, lane, wave);
      { const float c[4] = {C_A51, C_A52, C_A53, C_A54};
        combine<4>(y_s, k_s, c, hh, S0, nullptr, tid); }
      __syncthreads();
      f_eval(S0, S1, S2, V0L, vb0, V1L, vb1, V2h, vb2, gv, k_s + 4 * 2048, vst, lane, wave);
      { const float c[5] = {C_A61, C_A62, C_A63, C_A64, C_A65};
        combine<5>(y_s, k_s, c, hh, S0, nullptr, tid); }
      __syncthreads();
      f_eval(S0, S1, S2, V0L, vb0, V1L, vb1, V2h, vb2, gv, k_s + 5 * 2048, vst, lane, wave);
      { const float c[6] = {C_B1, C_B2, C_B3, C_B4, C_B5, C_B6};
        combine<6>(y_s, k_s, c, hh, S0, y_s, tid); }   // y update, restage f16
      __syncthreads();
    }

    if (gc) {
      for (int i = tid; i < 16 * HID; i += NTHREADS) {
        const int m = i >> 7, h = i & 127;
        out[((size_t)(bb + m) * T_STEPS + (t + 1)) * HID + h] = y_s[i];
      }
    } else if (t == T_STEPS - 2) {
      for (int i = tid; i < 16 * HID; i += NTHREADS) {
        const int m = i >> 7, h = i & 127;
        out[(size_t)(bb + m) * HID + h] = y_s[i];
      }
    }
    __syncthreads();
  }
}

extern "C" void kernel_launch(void* const* d_in, const int* in_sizes, int n_in,
                              void* d_out, int out_size, void* d_ws, size_t ws_size,
                              hipStream_t stream) {
  (void)in_sizes; (void)n_in; (void)out_size; (void)ws_size;
  const float* ts     = (const float*)d_in[0];
  const float* coeffs = (const float*)d_in[1];
  const float* W0 = (const float*)d_in[2];  const float* b0  = (const float*)d_in[3];
  const float* W1 = (const float*)d_in[4];  const float* b1  = (const float*)d_in[5];
  const float* W2 = (const float*)d_in[6];  const float* b2  = (const float*)d_in[7];
  const float* V0 = (const float*)d_in[8];  const float* vb0 = (const float*)d_in[9];
  const float* V1 = (const float*)d_in[10]; const float* vb1 = (const float*)d_in[11];
  const float* V2 = (const float*)d_in[12]; const float* vb2 = (const float*)d_in[13];
  const int*   gc = (const int*)d_in[14];
  _Float16* wsh = (_Float16*)d_ws;

  convert_weights<<<(N_TOTAL + 255) / 256, 256, 0, stream>>>(W0, W1, W2, V0, V1, V2, wsh);

  cde_kernel<<<8, NTHREADS, SMEM_BYTES, stream>>>(ts, coeffs, b0, b1, b2,
                                                  vb0, vb1, vb2, wsh, gc,
                                                  (float*)d_out);
}
